// MorphAwareTranslator_23553600651328
// MI455X (gfx1250) — matrix operand order repaired and checked by reading, not compiled, not run on hardware
//
#include <hip/hip_runtime.h>
#include <hip/hip_bf16.h>

// ---------------------------------------------------------------------------
// MorphAwareTranslator for MI455X (gfx1250): bf16 WMMA GEMM pipeline.
// GEMM is LDS-free and software-pipelined: stage s+1's six global b128
// fragment loads are issued before stage s's two WMMAs, so the wave waits
// with loadcnt<=6 instead of 0 and overlaps memory with the matrix pipe.
// ---------------------------------------------------------------------------

typedef __bf16 bf16_t;
typedef __attribute__((ext_vector_type(16))) __bf16 v16bf;
typedef __attribute__((ext_vector_type(8)))  __bf16 v8bf;
typedef __attribute__((ext_vector_type(8)))  float  v8f;
typedef unsigned int u32;

#define DEV __device__ __forceinline__

// Problem constants (match setup_inputs()).
constexpr int BB  = 32;     // batch
constexpr int SS  = 64;     // source length
constexpr int TT  = 64;     // target length
constexpr int EE  = 256;    // embed dim
constexpr int HH  = 512;    // hidden
constexpr int DIN = 512;    // E + 4*64
constexpr int G3H = 1536;   // 3*H
constexpr int VV  = 32000;  // vocab

DEV bf16_t f2bf(float f) {
    union { float f; u32 u; } v; v.f = f;
    u32 r = v.u + 0x7FFFu + ((v.u >> 16) & 1u);      // round-to-nearest-even
    union { unsigned short s; bf16_t b; } o; o.s = (unsigned short)(r >> 16);
    return o.b;
}
DEV float sigmoidf_(float x) { return 1.0f / (1.0f + __expf(-x)); }

// ---------------- Generic bf16 WMMA GEMM (LDS-free, pipelined) -------------
// C(M,N) f32 = A(M,K)bf16 @ Bpk [+ bias].  BM=32, BN=128, BK=32.
// grid = (N/128, M/32), block = 256 (8 waves).  Wave (wm,wn) computes the
// 16x32 C strip at rows wm*16, cols wn*32 (two 16x16 WMMA accumulators that
// share one A fragment).  Bpk is the fragment-native packed weight layout:
//   Bpk[((n>>4)*(K>>5) + (k>>5))*512 + (n&15)*32 + (k&31)]
// Callers guarantee M%32==0, N%128==0, K%32==0 with K/32 even, 16B-aligned A.
__global__ void k_gemm_bf16(const bf16_t* __restrict__ A, int lda,
                            const bf16_t* __restrict__ Bpk,
                            float* __restrict__ C, int ldc,
                            const float* __restrict__ bias, int K) {
    const int bn = blockIdx.x, bm = blockIdx.y;
    const int lane = threadIdx.x & 31, wave = threadIdx.x >> 5;
    const int wm = wave >> 2;          // 0..1 : M sub-tile
    const int wn = wave & 3;           // 0..3 : pair of N sub-tiles
    const int ml   = lane & 15;        // A row within tile / B col within tile
    const int half = lane >> 4;

    // A: lane reads row (bm*32 + wm*16 + ml), k-runs [half*8,+8) and +16 (ISA 7.12.2).
    const bf16_t* Ap = A + (size_t)(bm * 32 + wm * 16 + ml) * lda + half * 8;
    // B: two consecutive n-tiles; per-lane k-runs [half*8,+8) and +16 at n*32.
    const int nb0 = bn * 8 + wn * 2;
    const size_t nbstride = (size_t)(K >> 5) * 512;   // elements per n-tile block
    const bf16_t* Bp0 = Bpk + (size_t)nb0 * nbstride + ml * 32 + half * 8;
    const bf16_t* Bp1 = Bp0 + nbstride;

    union F { v16bf v; v8bf h[2]; };
    F aX, p0X, p1X, aY, p0Y, p1Y;

#define LOADF(AF, B0, B1, S)                                                  \
    AF.h[0] = *(const v8bf*)(Ap + ((S) << 5));                                \
    AF.h[1] = *(const v8bf*)(Ap + ((S) << 5) + 16);                           \
    B0.h[0] = *(const v8bf*)(Bp0 + (size_t)(S) * 512);                        \
    B0.h[1] = *(const v8bf*)(Bp0 + (size_t)(S) * 512 + 16);                   \
    B1.h[0] = *(const v8bf*)(Bp1 + (size_t)(S) * 512);                        \
    B1.h[1] = *(const v8bf*)(Bp1 + (size_t)(S) * 512 + 16);

#define WMMAF(AF, B0, B1)                                                     \
    acc0 = __builtin_amdgcn_wmma_f32_16x16x32_bf16(                           \
        false, AF.v, false, B0.v, (short)0, acc0, false, false);              \
    acc1 = __builtin_amdgcn_wmma_f32_16x16x32_bf16(                           \
        false, AF.v, false, B1.v, (short)0, acc1, false, false);

    v8f acc0 = {}, acc1 = {};
    const int K32 = K >> 5;            // even at every call site
    LOADF(aX, p0X, p1X, 0);
    int s = 0;
    for (; s + 2 < K32; s += 2) {
        LOADF(aY, p0Y, p1Y, s + 1);    // stage s+1 in flight while computing s
        WMMAF(aX, p0X, p1X);
        LOADF(aX, p0X, p1X, s + 2);    // stage s+2 in flight while computing s+1
        WMMAF(aY, p0Y, p1Y);
    }
    LOADF(aY, p0Y, p1Y, s + 1);        // last stage (K32-1)
    WMMAF(aX, p0X, p1X);
    WMMAF(aY, p0Y, p1Y);
#undef LOADF
#undef WMMAF

    // C/D layout: lane n = lane%16, VGPR r -> m = r + 8*(lane/16).
    const int m0  = bm * 32 + wm * 16 + half * 8;
    const int n0a = bn * 128 + wn * 32;
    float bv0 = bias ? bias[n0a + ml]      : 0.0f;
    float bv1 = bias ? bias[n0a + 16 + ml] : 0.0f;
#pragma unroll
    for (int r = 0; r < 8; ++r) {
        int m = m0 + r;
        C[(size_t)m * ldc + n0a + ml]      = acc0[r] + bv0;
        C[(size_t)m * ldc + n0a + 16 + ml] = acc1[r] + bv1;
    }
}

// ---------------- Weight pack: (N,K) f32 -> fragment-native bf16 -----------
// W is (N,K) f32 with row stride ldw starting at column col0.
__global__ void k_pack_b(const float* __restrict__ W, bf16_t* __restrict__ P,
                         int N, int K, int ldw, int col0) {
    int idx = blockIdx.x * 256 + threadIdx.x;
    if (idx >= N * K) return;
    int n = idx / K, k = idx - n * K;
    size_t off = ((size_t)(n >> 4) * (K >> 5) + (k >> 5)) * 512 + (n & 15) * 32 + (k & 31);
    P[off] = f2bf(W[(size_t)n * ldw + col0 + k]);
}

// ---------------- Gathers / packing ----------------------------------------
// Encoder input xs: (S,B,Din) bf16; padding_idx==0 rows are zero.
__global__ void k_build_enc_in(const int* __restrict__ xsurf,
                               const int* __restrict__ m0, const int* __restrict__ m1,
                               const int* __restrict__ m2, const int* __restrict__ m3,
                               const float* __restrict__ semb,
                               const float* __restrict__ me0, const float* __restrict__ me1,
                               const float* __restrict__ me2, const float* __restrict__ me3,
                               bf16_t* __restrict__ out) {
    int idx = blockIdx.x * 256 + threadIdx.x;       // S*B*512
    if (idx >= SS * BB * DIN) return;
    int d = idx & 511;
    int sb = idx >> 9;
    int b = sb & 31, s = sb >> 5;                   // row = s*B + b
    int bs = (b << 6) | s;                          // (B,S) index
    float v;
    if (d < EE) {
        int tk = xsurf[bs];
        v = tk ? semb[(size_t)tk * EE + d] : 0.0f;
    } else {
        int g = (d - EE) >> 6, dd = (d - EE) & 63;
        const float* tab = (g == 0) ? me0 : (g == 1) ? me1 : (g == 2) ? me2 : me3;
        const int*   mi  = (g == 0) ? m0  : (g == 1) ? m1  : (g == 2) ? m2  : m3;
        int tk = mi[bs];
        v = tk ? tab[(size_t)tk * 64 + dd] : 0.0f;
    }
    out[idx] = f2bf(v);
}

// yscat: (B*S, 2H) bf16 from ys_f/ys_b stored (S,B,H) f32.
__global__ void k_build_yscat(const float* __restrict__ ysf, const float* __restrict__ ysb,
                              bf16_t* __restrict__ out) {
    int idx = blockIdx.x * 256 + threadIdx.x;       // B*S*1024
    if (idx >= BB * SS * 2 * HH) return;
    int j = idx & (2 * HH - 1);
    int row = idx >> 10;
    int b = row >> 6, s = row & 63;
    float v = (j < HH) ? ysf[((size_t)s * BB + b) * HH + j]
                       : ysb[((size_t)s * BB + b) * HH + (j - HH)];
    out[idx] = f2bf(v);
}

__global__ void k_f32_to_bf16(const float* __restrict__ in, bf16_t* __restrict__ out, int n) {
    int idx = blockIdx.x * 256 + threadIdx.x;
    if (idx < n) out[idx] = f2bf(in[idx]);
}

__global__ void k_zero_h(float* __restrict__ h, bf16_t* __restrict__ hb) {
    int idx = blockIdx.x * 256 + threadIdx.x;
    if (idx >= BB * HH) return;
    h[idx] = 0.0f; hb[idx] = f2bf(0.0f);
}

// Decoder h0 = mean_s enc_out (B,S,H).
__global__ void k_mean_h(const float* __restrict__ enc_out,
                         float* __restrict__ h, bf16_t* __restrict__ hb) {
    int idx = blockIdx.x * 256 + threadIdx.x;       // B*H
    if (idx >= BB * HH) return;
    int b = idx / HH, j = idx - b * HH;
    float acc = 0.0f;
    for (int s = 0; s < SS; ++s) acc += enc_out[((size_t)b * SS + s) * HH + j];
    float m = acc * (1.0f / SS);
    h[idx] = m; hb[idx] = f2bf(m);
}

// GRU cell: r,z,n gate order; gx/gh rows have stride 3H and include biases.
__global__ void k_gru_cell(const float* __restrict__ gx, const float* __restrict__ gh,
                           const float* __restrict__ h_in, float* __restrict__ h_out,
                           bf16_t* __restrict__ h_out_bf, float* __restrict__ ys) {
    int idx = blockIdx.x * 256 + threadIdx.x;       // B*H
    if (idx >= BB * HH) return;
    int b = idx / HH, j = idx - b * HH;
    const float* gxr = gx + (size_t)b * G3H;
    const float* ghr = gh + (size_t)b * G3H;
    float r = sigmoidf_(gxr[j] + ghr[j]);
    float z = sigmoidf_(gxr[HH + j] + ghr[HH + j]);
    float n = tanhf(gxr[2 * HH + j] + r * ghr[2 * HH + j]);
    float hn = (1.0f - z) * n + z * h_in[idx];
    h_out[idx] = hn;
    h_out_bf[idx] = f2bf(hn);
    if (ys) ys[idx] = hn;
}

// Additive attention: scores/softmax/context.  Wh (B,H) includes attn_b;
// Wenc (B*S,H) is the precomputed enc-side projection.  One block per batch.
__global__ void k_attn(const float* __restrict__ Wh, const float* __restrict__ Wenc,
                       const float* __restrict__ enc_out, const float* __restrict__ vw,
                       float* __restrict__ context) {
    int b = blockIdx.x;
    int tid = threadIdx.x, lane = tid & 31, wave = tid >> 5;   // 8 waves
    __shared__ float sc[SS];
    for (int s = wave; s < SS; s += 8) {
        const float* wr = Wenc + ((size_t)b * SS + s) * HH;
        float part = 0.0f;
        for (int j = lane; j < HH; j += 32)
            part += tanhf(Wh[b * HH + j] + wr[j]) * vw[j];
#pragma unroll
        for (int off = 16; off; off >>= 1) part += __shfl_down(part, off, 32);
        if (lane == 0) sc[s] = part;
    }
    __syncthreads();
    if (tid == 0) {
        float mx = -1e30f;
        for (int s = 0; s < SS; ++s) mx = fmaxf(mx, sc[s]);
        float sum = 0.0f;
        for (int s = 0; s < SS; ++s) { float e = __expf(sc[s] - mx); sc[s] = e; sum += e; }
        float inv = 1.0f / sum;
        for (int s = 0; s < SS; ++s) sc[s] *= inv;
    }
    __syncthreads();
    for (int j = tid; j < HH; j += 256) {
        float acc = 0.0f;
        for (int s = 0; s < SS; ++s) acc += sc[s] * enc_out[((size_t)b * SS + s) * HH + j];
        context[b * HH + j] = acc;
    }
}

// Decoder GRU input xcat = [en_emb[y[:,t]], context] -> (B, E+H) bf16.
__global__ void k_build_xcat(const int* __restrict__ y, const float* __restrict__ en_emb,
                             const float* __restrict__ context, bf16_t* __restrict__ xcat,
                             int t) {
    int idx = blockIdx.x * 256 + threadIdx.x;       // B*768
    if (idx >= BB * (EE + HH)) return;
    int b = idx / (EE + HH), j = idx - b * (EE + HH);
    float v;
    if (j < EE) {
        int tok = y[b * TT + t];
        v = tok ? en_emb[(size_t)tok * EE + j] : 0.0f;
    } else {
        v = context[b * HH + (j - EE)];
    }
    xcat[idx] = f2bf(v);
}

// ---------------------------------------------------------------------------
extern "C" void kernel_launch(void* const* d_in, const int* in_sizes, int n_in,
                              void* d_out, int out_size, void* d_ws, size_t ws_size,
                              hipStream_t stream) {
    (void)in_sizes; (void)n_in; (void)out_size; (void)ws_size;
    const int*   x_surface = (const int*)d_in[0];
    const int*   m0 = (const int*)d_in[1];
    const int*   m1 = (const int*)d_in[2];
    const int*   m2 = (const int*)d_in[3];
    const int*   m3 = (const int*)d_in[4];
    const int*   y  = (const int*)d_in[5];
    const float* surface_emb = (const float*)d_in[6];
    const float* me0 = (const float*)d_in[7];
    const float* me1 = (const float*)d_in[8];
    const float* me2 = (const float*)d_in[9];
    const float* me3 = (const float*)d_in[10];
    const float* en_emb     = (const float*)d_in[11];
    const float* enc_Wih_f  = (const float*)d_in[12];
    const float* enc_Whh_f  = (const float*)d_in[13];
    const float* enc_bih_f  = (const float*)d_in[14];
    const float* enc_bhh_f  = (const float*)d_in[15];
    const float* enc_Wih_b  = (const float*)d_in[16];
    const float* enc_Whh_b  = (const float*)d_in[17];
    const float* enc_bih_b  = (const float*)d_in[18];
    const float* enc_bhh_b  = (const float*)d_in[19];
    const float* enc_proj_W = (const float*)d_in[20];
    const float* enc_proj_b = (const float*)d_in[21];
    const float* dec_Wih    = (const float*)d_in[22];
    const float* dec_Whh    = (const float*)d_in[23];
    const float* dec_bih    = (const float*)d_in[24];
    const float* dec_bhh    = (const float*)d_in[25];
    const float* fc_W       = (const float*)d_in[26];
    const float* fc_b       = (const float*)d_in[27];
    const float* attn_W     = (const float*)d_in[28];
    const float* attn_b     = (const float*)d_in[29];
    const float* v_w        = (const float*)d_in[30];
    float* out = (float*)d_out;

    // ---- workspace carve-up (256B aligned) ----
    char* wp = (char*)d_ws;
    auto alloc = [&](size_t bytes) -> void* {
        void* r = (void*)wp; wp += (bytes + 255) & ~(size_t)255; return r;
    };
    bf16_t* Wt_ih_f  = (bf16_t*)alloc((size_t)DIN * G3H * 2);
    bf16_t* Wt_hh_f  = (bf16_t*)alloc((size_t)HH * G3H * 2);
    bf16_t* Wt_ih_b  = (bf16_t*)alloc((size_t)DIN * G3H * 2);
    bf16_t* Wt_hh_b  = (bf16_t*)alloc((size_t)HH * G3H * 2);
    bf16_t* Wt_proj  = (bf16_t*)alloc((size_t)2 * HH * HH * 2);
    bf16_t* Wt_dih   = (bf16_t*)alloc((size_t)(EE + HH) * G3H * 2);
    bf16_t* Wt_dhh   = (bf16_t*)alloc((size_t)HH * G3H * 2);
    bf16_t* Wt_fc    = (bf16_t*)alloc((size_t)HH * VV * 2);
    bf16_t* Wt_ah    = (bf16_t*)alloc((size_t)HH * HH * 2);
    bf16_t* Wt_ae    = (bf16_t*)alloc((size_t)HH * HH * 2);
    bf16_t* xs       = (bf16_t*)alloc((size_t)SS * BB * DIN * 2);
    float*  gx_f     = (float*)alloc((size_t)SS * BB * G3H * 4);
    float*  gx_b     = (float*)alloc((size_t)SS * BB * G3H * 4);
    float*  ys_f     = (float*)alloc((size_t)SS * BB * HH * 4);
    float*  ys_b     = (float*)alloc((size_t)SS * BB * HH * 4);
    bf16_t* yscat    = (bf16_t*)alloc((size_t)BB * SS * 2 * HH * 2);
    float*  enc_out  = (float*)alloc((size_t)BB * SS * HH * 4);
    bf16_t* enc_out_bf = (bf16_t*)alloc((size_t)BB * SS * HH * 2);
    float*  Wenc     = (float*)alloc((size_t)BB * SS * HH * 4);
    float*  h_f      = (float*)alloc((size_t)BB * HH * 4);
    bf16_t* h_f_bf   = (bf16_t*)alloc((size_t)BB * HH * 2);
    float*  h_b      = (float*)alloc((size_t)BB * HH * 4);
    bf16_t* h_b_bf   = (bf16_t*)alloc((size_t)BB * HH * 2);
    float*  h_d      = (float*)alloc((size_t)BB * HH * 4);
    bf16_t* h_d_bf   = (bf16_t*)alloc((size_t)BB * HH * 2);
    float*  gh       = (float*)alloc((size_t)BB * G3H * 4);
    float*  gx_dec   = (float*)alloc((size_t)BB * G3H * 4);
    float*  Wh       = (float*)alloc((size_t)BB * HH * 4);
    float*  context  = (float*)alloc((size_t)BB * HH * 4);
    bf16_t* xcat     = (bf16_t*)alloc((size_t)BB * (EE + HH) * 2);

    auto tgrid = [](int n) { return dim3((n + 255) / 256); };
    auto PACKB = [&](const float* W, bf16_t* P, int N, int K, int ldw, int col0) {
        k_pack_b<<<tgrid(N * K), 256, 0, stream>>>(W, P, N, K, ldw, col0);
    };
    auto GEMM = [&](const bf16_t* A, int lda, const bf16_t* Bpk,
                    float* C, int ldc, const float* bias, int M, int N, int K) {
        k_gemm_bf16<<<dim3(N / 128, M / 32), 256, 0, stream>>>(A, lda, Bpk, C, ldc, bias, K);
    };

    // ---- weights -> bf16, fragment-native packing ----
    PACKB(enc_Wih_f,  Wt_ih_f, G3H, DIN, DIN, 0);
    PACKB(enc_Whh_f,  Wt_hh_f, G3H, HH,  HH,  0);
    PACKB(enc_Wih_b,  Wt_ih_b, G3H, DIN, DIN, 0);
    PACKB(enc_Whh_b,  Wt_hh_b, G3H, HH,  HH,  0);
    PACKB(enc_proj_W, Wt_proj, HH, 2 * HH, 2 * HH, 0);
    PACKB(dec_Wih,    Wt_dih,  G3H, EE + HH, EE + HH, 0);
    PACKB(dec_Whh,    Wt_dhh,  G3H, HH,  HH,  0);
    PACKB(fc_W,       Wt_fc,   VV,  HH,  HH,  0);
    PACKB(attn_W,     Wt_ah,   HH,  HH,  2 * HH, 0);    // h-side columns [0,H)
    PACKB(attn_W,     Wt_ae,   HH,  HH,  2 * HH, HH);   // enc-side columns [H,2H)

    // ---- encoder inputs + hoisted input projections ----
    k_build_enc_in<<<tgrid(SS * BB * DIN), 256, 0, stream>>>(
        x_surface, m0, m1, m2, m3, surface_emb, me0, me1, me2, me3, xs);
    GEMM(xs, DIN, Wt_ih_f, gx_f, G3H, enc_bih_f, SS * BB, G3H, DIN);
    GEMM(xs, DIN, Wt_ih_b, gx_b, G3H, enc_bih_b, SS * BB, G3H, DIN);

    // ---- encoder recurrences ----
    k_zero_h<<<tgrid(BB * HH), 256, 0, stream>>>(h_f, h_f_bf);
    for (int s = 0; s < SS; ++s) {
        GEMM(h_f_bf, HH, Wt_hh_f, gh, G3H, enc_bhh_f, BB, G3H, HH);
        k_gru_cell<<<tgrid(BB * HH), 256, 0, stream>>>(
            gx_f + (size_t)s * BB * G3H, gh, h_f, h_f, h_f_bf, ys_f + (size_t)s * BB * HH);
    }
    k_zero_h<<<tgrid(BB * HH), 256, 0, stream>>>(h_b, h_b_bf);
    for (int i = 0; i < SS; ++i) {
        int s = SS - 1 - i;
        GEMM(h_b_bf, HH, Wt_hh_b, gh, G3H, enc_bhh_b, BB, G3H, HH);
        k_gru_cell<<<tgrid(BB * HH), 256, 0, stream>>>(
            gx_b + (size_t)s * BB * G3H, gh, h_b, h_b, h_b_bf, ys_b + (size_t)s * BB * HH);
    }

    // ---- enc projection + precomputed attention enc-side ----
    k_build_yscat<<<tgrid(BB * SS * 2 * HH), 256, 0, stream>>>(ys_f, ys_b, yscat);
    GEMM(yscat, 2 * HH, Wt_proj, enc_out, HH, enc_proj_b, BB * SS, HH, 2 * HH);
    k_f32_to_bf16<<<tgrid(BB * SS * HH), 256, 0, stream>>>(enc_out, enc_out_bf, BB * SS * HH);
    GEMM(enc_out_bf, HH, Wt_ae, Wenc, HH, nullptr, BB * SS, HH, HH);
    k_mean_h<<<tgrid(BB * HH), 256, 0, stream>>>(enc_out, h_d, h_d_bf);

    // ---- decoder ----
    const int LDOUT = (TT - 1) * VV;                 // row stride of (B, T-1, V)
    for (int t = 0; t < TT - 1; ++t) {
        GEMM(h_d_bf, HH, Wt_ah, Wh, HH, attn_b, BB, HH, HH);
        k_attn<<<dim3(BB), 256, 0, stream>>>(Wh, Wenc, enc_out, v_w, context);
        k_build_xcat<<<tgrid(BB * (EE + HH)), 256, 0, stream>>>(y, en_emb, context, xcat, t);
        GEMM(xcat, EE + HH, Wt_dih, gx_dec, G3H, dec_bih, BB, G3H, EE + HH);
        GEMM(h_d_bf, HH, Wt_dhh, gh, G3H, dec_bhh, BB, G3H, HH);
        k_gru_cell<<<tgrid(BB * HH), 256, 0, stream>>>(gx_dec, gh, h_d, h_d, h_d_bf, nullptr);
        GEMM(h_d_bf, HH, Wt_fc, out + (size_t)t * VV, LDOUT, fc_b, BB, VV, HH);
    }
}
